// MultiHopAttGNN_15693810500054
// MI455X (gfx1250) — matrix operand order repaired and checked by reading, not compiled, not run on hardware
//
#include <hip/hip_runtime.h>
#include <hip/hip_bf16.h>

// ---------------------------------------------------------------------------
// MultiHopAttGNN on MI455X (gfx1250, wave32).
// x@W GEMM: f16 staged in LDS (converted during staging), contiguous
// layout-exact fragments (ds_load_b128), v_wmma_f32_16x16x32_f16 inner loop,
// async global->LDS DMA (global_load_async_to_lds_b128) for the weight tile.
// ---------------------------------------------------------------------------

typedef __attribute__((ext_vector_type(16))) _Float16 v16h;
typedef __attribute__((ext_vector_type(8)))  _Float16 v8h;
typedef __attribute__((ext_vector_type(8)))  float    v8f;
typedef __attribute__((__vector_size__(4 * sizeof(int)))) int v4i;

#define FIN 1024
#define DH  128

#if defined(__has_builtin)
#  if __has_builtin(__builtin_amdgcn_global_load_async_to_lds_b128)
#    define HAVE_ASYNC_LDS 1
#  else
#    define HAVE_ASYNC_LDS 0
#  endif
#  if __has_builtin(__builtin_amdgcn_s_wait_asynccnt)
#    define HAVE_WAIT_ASYNC 1
#  else
#    define HAVE_WAIT_ASYNC 0
#  endif
#else
#  define HAVE_ASYNC_LDS 0
#  define HAVE_WAIT_ASYNC 0
#endif

typedef __attribute__((address_space(1))) v4i* gv4p;   // global int4*
typedef __attribute__((address_space(3))) v4i* lv4p;   // LDS int4*

// ---- monotonic float<->uint key for atomicMax-based segment_max ------------
__device__ __forceinline__ unsigned fkey(float f) {
  unsigned u = __float_as_uint(f);
  return (u & 0x80000000u) ? ~u : (u | 0x80000000u);
}
__device__ __forceinline__ float funkey(unsigned k) {
  unsigned u = (k & 0x80000000u) ? (k & 0x7fffffffu) : ~k;
  return __uint_as_float(u);
}

// ===========================================================================
// W[1024,128] fp32  ->  WT[n][k] f16 (transposed, row-major over k)
// ===========================================================================
__global__ void convert_w_t(const float* __restrict__ W, _Float16* __restrict__ WT) {
  int g = blockIdx.x * blockDim.x + threadIdx.x;   // n*FIN + k
  if (g >= FIN * DH) return;
  int n = g >> 10;          // / FIN
  int k = g & (FIN - 1);
  WT[(size_t)n * FIN + k] = (_Float16)W[(size_t)k * DH + n];
}

// ===========================================================================
// GEMM: H[N,128] = X[N,1024] @ W[1024,128]   (f16 WMMA, fp32 acc)
// Block = 256 threads (8 waves). Block tile: 128 rows x 128 cols.
// Wave w owns rows [w*16, w*16+16) x all 128 cols (8 accumulators).
// LDS holds f16 tiles:  sA[m][k] (x, converted on store),
//                       sBT[n][k] (pre-converted W, raw async copy).
// ===========================================================================
__global__ __launch_bounds__(256) void gemm_xw_wmma(
    const float* __restrict__ X, const _Float16* __restrict__ WT,
    float* __restrict__ H, int nrows) {
  __shared__ __attribute__((aligned(16))) _Float16 sA [128][32];  // 8 KB
  __shared__ __attribute__((aligned(16))) _Float16 sBT[128][32];  // 8 KB

  const int tid  = threadIdx.x;
  const int wave = tid >> 5;
  const int lane = tid & 31;
  const int hl   = lane >> 4;     // half-wave select
  const int lm   = lane & 15;
  const int row0 = blockIdx.x * 128;

  // B staging assignment: thread t copies 32 bytes of row n = t>>1
  const int bn   = tid >> 1;
  const int bp   = tid & 1;       // which 16-f16 half of the 32-wide row

  v8f acc[8] = {};

  for (int k0 = 0; k0 < FIN; k0 += 32) {
    __syncthreads();              // previous iteration's fragment reads done

    // --- stage A tile: 128x32 fp32 -> f16, 4 float4 per thread ---
    #pragma unroll
    for (int i = 0; i < 4; ++i) {
      int el = (tid + i * 256) * 4;   // 0..4095
      int r  = el >> 5;               // /32
      int k  = el & 31;
      int row = row0 + r;
      float4 v = make_float4(0.f, 0.f, 0.f, 0.f);
      if (row < nrows)
        v = *(const float4*)(X + (size_t)row * FIN + k0 + k);
      _Float16 h4[4] = {(_Float16)v.x, (_Float16)v.y, (_Float16)v.z, (_Float16)v.w};
      *(uint2*)&sA[r][k] = *(uint2*)h4;
    }

    // --- stage B tile: 128x32 f16, raw copy of WT rows ---
    {
      const _Float16* gsrc = WT + (size_t)bn * FIN + k0 + bp * 16;
      _Float16*       ldst = &sBT[bn][bp * 16];
#if HAVE_ASYNC_LDS
      __builtin_amdgcn_global_load_async_to_lds_b128(
          (gv4p)(gsrc), (lv4p)(ldst), 0, 0);
      __builtin_amdgcn_global_load_async_to_lds_b128(
          (gv4p)(gsrc), (lv4p)(ldst), 16, 0);
#else
      *(uint4*)(ldst + 0) = *(const uint4*)(gsrc + 0);
      *(uint4*)(ldst + 8) = *(const uint4*)(gsrc + 8);
#endif
    }

    // prefetch next x chunk (global_prefetch_b8)
    if (k0 + 32 < FIN) {
      int row = row0 + (tid >> 1);
      if (row < nrows)
        __builtin_prefetch(X + (size_t)row * FIN + k0 + 32 + (tid & 1) * 16, 0, 0);
    }

#if HAVE_ASYNC_LDS
#  if HAVE_WAIT_ASYNC
    __builtin_amdgcn_s_wait_asynccnt(0);
#  else
    asm volatile("s_wait_asynccnt 0x0" ::: "memory");
#  endif
#endif
    __syncthreads();

    // --- A fragment: contiguous per documented 16-bit A layout:
    //     v16h elems 0..7  = K (hl*8 .. hl*8+7)
    //     v16h elems 8..15 = K (16+hl*8 .. 16+hl*8+7)
    const int m = wave * 16 + lm;
    v8h alo = *(const v8h*)&sA[m][hl * 8];
    v8h ahi = *(const v8h*)&sA[m][16 + hl * 8];
    v16h afrag = __builtin_shufflevector(alo, ahi,
        0, 1, 2, 3, 4, 5, 6, 7, 8, 9, 10, 11, 12, 13, 14, 15);

    // --- 8 B fragments: elems 0..7 = K (hl*8 ..), elems 8..15 = K (16+hl*8 ..) for column n
    #pragma unroll
    for (int j = 0; j < 8; ++j) {
      const int n = j * 16 + lm;
      v8h blo = *(const v8h*)&sBT[n][hl * 8];
      v8h bhi = *(const v8h*)&sBT[n][16 + hl * 8];
      v16h bfrag = __builtin_shufflevector(blo, bhi,
          0, 1, 2, 3, 4, 5, 6, 7, 8, 9, 10, 11, 12, 13, 14, 15);
      acc[j] = __builtin_amdgcn_wmma_f32_16x16x32_f16(
          false, afrag, false, bfrag, (short)0, acc[j], false, false);
    }
  }

  // --- store: C VGPR r -> (M = r + hl*8, N = lm) within 16x16 tile ---
  const int mbase = row0 + wave * 16 + hl * 8;
  #pragma unroll
  for (int j = 0; j < 8; ++j) {
    const int col = j * 16 + lm;
    #pragma unroll
    for (int r = 0; r < 8; ++r) {
      int row = mbase + r;
      if (row < nrows) H[(size_t)row * DH + col] = acc[j][r];
    }
  }
}

// ===========================================================================
// al_s[i] = h[i]·a_src ; al_d[i] = h[i]·a_dst
// ===========================================================================
__global__ void compute_al(const float* __restrict__ H,
                           const float* __restrict__ asrc,
                           const float* __restrict__ adst,
                           float* __restrict__ als, float* __restrict__ ald,
                           int n) {
  int i = blockIdx.x * blockDim.x + threadIdx.x;
  if (i >= n) return;
  const float* h = H + (size_t)i * DH;
  float s = 0.f, d = 0.f;
  #pragma unroll 4
  for (int k = 0; k < DH; k += 4) {
    float4 hv = *(const float4*)(h + k);
    float4 av = *(const float4*)(asrc + k);
    float4 dv = *(const float4*)(adst + k);
    s += hv.x * av.x + hv.y * av.y + hv.z * av.z + hv.w * av.w;
    d += hv.x * dv.x + hv.y * dv.y + hv.z * dv.z + hv.w * dv.w;
  }
  als[i] = s; ald[i] = d;
}

// ===========================================================================
// init / fill helpers (grid-stride)
// ===========================================================================
__global__ void fill_zero(float* p, size_t n) {
  size_t i  = blockIdx.x * (size_t)blockDim.x + threadIdx.x;
  size_t st = (size_t)gridDim.x * blockDim.x;
  for (; i < n; i += st) p[i] = 0.f;
}

__global__ void init_hop(float* outh, unsigned* mkey, float* denom,
                         int n, size_t nd) {
  size_t i  = blockIdx.x * (size_t)blockDim.x + threadIdx.x;
  size_t st = (size_t)gridDim.x * blockDim.x;
  unsigned negk = fkey(-3.402823466e38f);
  for (size_t j = i; j < nd; j += st) outh[j] = 0.f;
  for (size_t j = i; j < (size_t)n; j += st) { mkey[j] = negk; denom[j] = 0.f; }
}

// ===========================================================================
// edge phase (edges 0..E-1 from ei, edges E..E+N-1 are self loops)
// ===========================================================================
__global__ void edge_logits(const int* __restrict__ ei,
                            const float* __restrict__ als,
                            const float* __restrict__ ald,
                            float* __restrict__ ebuf, unsigned* __restrict__ mkey,
                            int E_, int ME_) {
  int i = blockIdx.x * blockDim.x + threadIdx.x;
  if (i >= ME_) return;
  int s, d;
  if (i < E_) { s = ei[i]; d = ei[E_ + i]; } else { s = d = i - E_; }
  float lg = als[s] + ald[d];
  lg = lg > 0.f ? lg : 0.2f * lg;          // GAT leaky_relu slope 0.2
  ebuf[i] = lg;
  atomicMax(&mkey[d], fkey(lg));
}

__global__ void edge_exp(const int* __restrict__ ei,
                         float* __restrict__ ebuf,
                         const unsigned* __restrict__ mkey,
                         float* __restrict__ denom, int E_, int ME_) {
  int i = blockIdx.x * blockDim.x + threadIdx.x;
  if (i >= ME_) return;
  int d = (i < E_) ? ei[E_ + i] : (i - E_);
  float ex = __expf(ebuf[i] - funkey(mkey[d]));
  ebuf[i] = ex;
  atomicAdd(&denom[d], ex);
}

// one wave per edge: lane handles 4 of the 128 features
__global__ void edge_scatter(const int* __restrict__ ei,
                             const float* __restrict__ ebuf,
                             const float* __restrict__ denom,
                             const float* __restrict__ H,
                             float* __restrict__ outh, int E_, int ME_) {
  long long g = (long long)blockIdx.x * blockDim.x + threadIdx.x;
  int edge = (int)(g >> 5);
  int lane = (int)(g & 31);
  if (edge >= ME_) return;
  int s, d;
  if (edge < E_) { s = ei[edge]; d = ei[E_ + edge]; } else { s = d = edge - E_; }
  float alpha = ebuf[edge] / denom[d];
  float4 hv = *(const float4*)(H + (size_t)s * DH + lane * 4);
  float* o = outh + (size_t)d * DH + lane * 4;
  atomicAdd(o + 0, alpha * hv.x);
  atomicAdd(o + 1, alpha * hv.y);
  atomicAdd(o + 2, alpha * hv.z);
  atomicAdd(o + 3, alpha * hv.w);
}

__global__ void acc_lrelu(float* __restrict__ acc, const float* __restrict__ outh,
                          const float* __restrict__ bb, size_t nd) {
  size_t i  = blockIdx.x * (size_t)blockDim.x + threadIdx.x;
  size_t st = (size_t)gridDim.x * blockDim.x;
  for (size_t j = i; j < nd; j += st) {
    float v = outh[j] + bb[j & (DH - 1)];
    acc[j] += v > 0.f ? v : 0.01f * v;
  }
}

// ===========================================================================
// pooling
// ===========================================================================
__global__ void pool_scatter(const float* __restrict__ acc,
                             const int* __restrict__ batch,
                             float* __restrict__ pooled,
                             float* __restrict__ counts, int n) {
  long long g = (long long)blockIdx.x * blockDim.x + threadIdx.x;
  int node = (int)(g >> 5);
  int lane = (int)(g & 31);
  if (node >= n) return;
  int b = batch[node];
  if (lane == 0) atomicAdd(&counts[b], 1.0f);
  float4 v = *(const float4*)(acc + (size_t)node * DH + lane * 4);
  float* p = pooled + (size_t)b * DH + lane * 4;
  atomicAdd(p + 0, v.x);
  atomicAdd(p + 1, v.y);
  atomicAdd(p + 2, v.z);
  atomicAdd(p + 3, v.w);
}

__global__ void pool_div(float* pooled, const float* counts, int bd) {
  int i = blockIdx.x * blockDim.x + threadIdx.x;
  if (i >= bd) return;
  pooled[i] /= fmaxf(counts[i >> 7], 1.0f);
}

// ===========================================================================
// small dense layers: Y[r, c] = act(X[r,:K] . W[:,c] + b[c]) with row stride
// act: 0 = leaky_relu(0.01), 1 = sigmoid
// ===========================================================================
__global__ void dense(const float* __restrict__ X, const float* __restrict__ Wm,
                      const float* __restrict__ bias, float* __restrict__ Y,
                      int R, int K, int C, int ldY, int act) {
  int g = blockIdx.x * blockDim.x + threadIdx.x;
  if (g >= R * C) return;
  int r = g / C, c = g % C;
  float s = bias[c];
  for (int k = 0; k < K; ++k)
    s += X[(size_t)r * K + k] * Wm[(size_t)k * C + c];
  if (act == 0)      s = s > 0.f ? s : 0.01f * s;
  else if (act == 1) s = 1.0f / (1.0f + __expf(-s));
  Y[(size_t)r * ldY + c] = s;
}

// ===========================================================================
// launch
// ===========================================================================
extern "C" void kernel_launch(void* const* d_in, const int* in_sizes, int n_in,
                              void* d_out, int out_size, void* d_ws, size_t ws_size,
                              hipStream_t stream) {
  const float* p1x  = (const float*)d_in[0];
  const int*   p1e1 = (const int*)d_in[1];
  const int*   p1e2 = (const int*)d_in[2];
  const int*   p1e3 = (const int*)d_in[3];
  const int*   p1b  = (const int*)d_in[4];
  const float* p2x  = (const float*)d_in[5];
  const int*   p2e1 = (const int*)d_in[6];
  const int*   p2e2 = (const int*)d_in[7];
  const int*   p2e3 = (const int*)d_in[8];
  const int*   p2b  = (const int*)d_in[9];
  const float* W1    = (const float*)d_in[10];
  const float* a1s   = (const float*)d_in[11];
  const float* a1d   = (const float*)d_in[12];
  const float* b1    = (const float*)d_in[13];
  const float* p1fcW = (const float*)d_in[14];
  const float* p1fcb = (const float*)d_in[15];
  const float* W2    = (const float*)d_in[16];
  const float* a2s   = (const float*)d_in[17];
  const float* a2d   = (const float*)d_in[18];
  const float* b2    = (const float*)d_in[19];
  const float* p2fcW = (const float*)d_in[20];
  const float* p2fcb = (const float*)d_in[21];
  const float* fc1W  = (const float*)d_in[22];
  const float* fc1b  = (const float*)d_in[23];
  const float* fc2W  = (const float*)d_in[24];
  const float* fc2b  = (const float*)d_in[25];
  const float* outW  = (const float*)d_in[26];
  const float* outB  = (const float*)d_in[27];

  const int N  = in_sizes[4];
  const int E  = in_sizes[1] / 2;
  const int B  = out_size;            // output is [B, 1]
  const int ME = E + N;               // edges + self loops
  const size_t ND = (size_t)N * DH;
  (void)n_in; (void)ws_size;

  // ---- workspace carve-up (float units) ----
  float* ws = (float*)d_ws;
  size_t off = 0;
  float*    h      = ws + off; off += ND;
  float*    acc    = ws + off; off += ND;
  float*    outh   = ws + off; off += ND;
  float*    als    = ws + off; off += N;
  float*    ald    = ws + off; off += N;
  float*    ebuf   = ws + off; off += ME;
  unsigned* mkey   = (unsigned*)(ws + off); off += N;
  float*    denom  = ws + off; off += N;
  float*    pooled = ws + off; off += (size_t)B * DH;
  float*    counts = ws + off; off += B;   // contiguous with pooled
  float*    xc     = ws + off; off += (size_t)B * 256;
  float*    y1     = ws + off; off += (size_t)B * 256;
  float*    y2     = ws + off; off += (size_t)B * 64;
  _Float16* wth    = (_Float16*)(ws + off); off += (size_t)FIN * DH / 2;

  struct BranchArgs {
    const float* x; const int* e[3]; const int* bat;
    const float *W, *as_, *ad_, *bb, *fw, *fb; int xcoff;
  } br[2] = {
    { p1x, {p1e1, p1e2, p1e3}, p1b, W1, a1s, a1d, b1, p1fcW, p1fcb, 0  },
    { p2x, {p2e1, p2e2, p2e3}, p2b, W2, a2s, a2d, b2, p2fcW, p2fcb, DH },
  };

  const int gemm_blocks = (N + 127) / 128;
  const int edge_blocks = (ME + 255) / 256;
  const int scat_blocks = (int)(((long long)ME * 32 + 255) / 256);
  const int pools_blocks = (int)(((long long)N * 32 + 255) / 256);

  for (int bi = 0; bi < 2; ++bi) {
    const BranchArgs& P = br[bi];
    convert_w_t<<<(FIN * DH + 255) / 256, 256, 0, stream>>>(P.W, wth);
    gemm_xw_wmma<<<gemm_blocks, 256, 0, stream>>>(P.x, wth, h, N);
    compute_al<<<(N + 255) / 256, 256, 0, stream>>>(h, P.as_, P.ad_, als, ald, N);
    fill_zero<<<1024, 256, 0, stream>>>(acc, ND);
    for (int hop = 0; hop < 3; ++hop) {
      init_hop<<<1024, 256, 0, stream>>>(outh, mkey, denom, N, ND);
      edge_logits<<<edge_blocks, 256, 0, stream>>>(P.e[hop], als, ald, ebuf, mkey, E, ME);
      edge_exp<<<edge_blocks, 256, 0, stream>>>(P.e[hop], ebuf, mkey, denom, E, ME);
      edge_scatter<<<scat_blocks, 256, 0, stream>>>(P.e[hop], ebuf, denom, h, outh, E, ME);
      acc_lrelu<<<1024, 256, 0, stream>>>(acc, outh, P.bb, ND);
    }
    fill_zero<<<64, 256, 0, stream>>>(pooled, (size_t)B * DH + B); // pooled+counts
    pool_scatter<<<pools_blocks, 256, 0, stream>>>(acc, P.bat, pooled, counts, N);
    pool_div<<<(B * DH + 255) / 256, 256, 0, stream>>>(pooled, counts, B * DH);
    dense<<<(B * DH + 255) / 256, 256, 0, stream>>>(pooled, P.fw, P.fb,
                                                    xc + P.xcoff, B, DH, DH, 256, 0);
  }

  dense<<<(B * 256 + 255) / 256, 256, 0, stream>>>(xc, fc1W, fc1b, y1, B, 256, 256, 256, 0);
  dense<<<(B * 64 + 255) / 256, 256, 0, stream>>>(y1, fc2W, fc2b, y2, B, 256, 64, 64, 0);
  dense<<<(B + 255) / 256, 256, 0, stream>>>(y2, outW, outB, (float*)d_out, B, 64, 1, 1, 1);
}
